// FiLM_Conditioning_Block_38182259261866
// MI455X (gfx1250) — compile-verified, matrix operand order checked by reading, not run on hardware
//
#include <hip/hip_runtime.h>
#include <hip/hip_bf16.h>

// ---------------- problem constants ----------------
#define kNT   32768   // N*T rows
#define kCIN  64
#define kH    128
#define kO    16
#define kF2   16
#define kRS   130     // LDS row stride in bf16 elements (128 + 2 pad -> conflict-free)

// per-o packed weight block: 52 tiles (16 L0 + 32 L1 + 4 L2) x 1024 bytes
#define kTILES_O   52
#define kOBLK_B    (kTILES_O * 1024)          // 53248 bytes per o
#define kOBLK_U    (kOBLK_B / 4)              // 13312 uints per o

typedef __attribute__((ext_vector_type(16))) __bf16 v16bf;
typedef __attribute__((ext_vector_type(8)))  float  v8f;

struct U32x8 { uint4 a, b; };   // 32 bytes == one v16bf operand

static __device__ __forceinline__ unsigned pack_bf16(float a, float b) {
    unsigned short lo = __builtin_bit_cast(unsigned short, __float2bfloat16(a));
    unsigned short hi = __builtin_bit_cast(unsigned short, __float2bfloat16(b));
    return (unsigned)lo | ((unsigned)hi << 16);
}
static __device__ __forceinline__ unsigned short bf_bits(float a) {
    return __builtin_bit_cast(unsigned short, __float2bfloat16(a));
}
// 16-bit A/B operand lane layout: half-pair v (v=0..7) holds K = fmap(v)+[0,1] (+8 for lanes 16..31)
static __device__ __forceinline__ int fmap(int v) { return (v < 4) ? 2 * v : 2 * v + 8; }
static __device__ __forceinline__ float leaky(float v) { return v >= 0.f ? v : 0.01f * v; }

// lane-slot counts for the pack kernel
#define kC0 (kO * 2 * 8 * 32)   // 8192  : W0, 2 k-chunks, 8 n-tiles
#define kC1 (kO * 4 * 8 * 32)   // 16384 : W1, 4 k-chunks, 8 n-tiles
#define kC2 (kO * 4 * 1 * 32)   // 2048  : W2, 4 k-chunks, 1 n-tile

// ---------------- weight pre-pack: f32 -> bf16, per-o contiguous, B-operand lane layout ----------------
__global__ void film_pack_weights(const float* __restrict__ W0,
                                  const float* __restrict__ W1,
                                  const float* __restrict__ W2,
                                  unsigned* __restrict__ ws) {
    int idx = blockIdx.x * blockDim.x + threadIdx.x;

    if (idx < kC0) {
        // B[k=c][n=h] = W0[o, h, c]   -> tile index (kc*8 + nt) within o-block
        int lane = idx & 31, nt = (idx >> 5) & 7, kc = (idx >> 8) & 1, o = idx >> 9;
        int ng = nt * 16 + (lane & 15);
        int kh = 8 * (lane >> 4);
        const float* src = W0 + ((size_t)o * kH + ng) * kCIN;
        unsigned* dst = ws + (size_t)o * kOBLK_U + (kc * 8 + nt) * 256 + lane * 8;
#pragma unroll
        for (int v = 0; v < 8; ++v) {
            int k = kc * 32 + fmap(v) + kh;
            float2 w = *(const float2*)(src + k);
            dst[v] = pack_bf16(w.x, w.y);
        }
    } else if (idx < kC0 + kC1) {
        // B[k=h][n=g] = W1[o, g, h]   -> tile index 16 + (kc*8 + nt)
        int j = idx - kC0;
        int lane = j & 31, nt = (j >> 5) & 7, kc = (j >> 8) & 3, o = j >> 10;
        int ng = nt * 16 + (lane & 15);
        int kh = 8 * (lane >> 4);
        const float* src = W1 + ((size_t)o * kH + ng) * kH;
        unsigned* dst = ws + (size_t)o * kOBLK_U + (16 + kc * 8 + nt) * 256 + lane * 8;
#pragma unroll
        for (int v = 0; v < 8; ++v) {
            int k = kc * 32 + fmap(v) + kh;
            float2 w = *(const float2*)(src + k);
            dst[v] = pack_bf16(w.x, w.y);
        }
    } else if (idx < kC0 + kC1 + kC2) {
        // B[k=h][n=f] = W2[o, f, h]   -> tile index 48 + kc
        int j = idx - kC0 - kC1;
        int lane = j & 31, kc = (j >> 5) & 3, o = j >> 7;
        int ng = lane & 15;
        int kh = 8 * (lane >> 4);
        const float* src = W2 + ((size_t)o * kF2 + ng) * kH;
        unsigned* dst = ws + (size_t)o * kOBLK_U + (48 + kc) * 256 + lane * 8;
#pragma unroll
        for (int v = 0; v < 8; ++v) {
            int k = kc * 32 + fmap(v) + kh;
            float2 w = *(const float2*)(src + k);
            dst[v] = pack_bf16(w.x, w.y);
        }
    }
}

// ---------------- bias + leaky + LayerNorm, result staged to LDS as bf16 ----------------
static __device__ __forceinline__ void bias_leaky_ln_store(
        v8f* acc, const float* __restrict__ bias, const float* __restrict__ gamma,
        const float* __restrict__ beta, int o, int col, int hi, unsigned short* sh) {
#pragma unroll
    for (int nt = 0; nt < 8; ++nt) {
        float b = bias[o * kH + nt * 16 + col];
#pragma unroll
        for (int r = 0; r < 8; ++r) acc[nt][r] = leaky(acc[nt][r] + b);
    }
    // LN over H=128 per row m = r + 8*hi. Row m lives only in the 16-lane half 'hi',
    // so xor-shuffles with masks 1,2,4,8 reduce exactly across that row's lanes.
    float nrs[8], nmu[8];
#pragma unroll
    for (int r = 0; r < 8; ++r) {
        float s = 0.f, q = 0.f;
#pragma unroll
        for (int nt = 0; nt < 8; ++nt) { float v = acc[nt][r]; s += v; q += v * v; }
#pragma unroll
        for (int m = 1; m <= 8; m <<= 1) {
            s += __shfl_xor(s, m, 32);
            q += __shfl_xor(q, m, 32);
        }
        float mean = s * (1.f / 128.f);
        float var  = q * (1.f / 128.f) - mean * mean;
        float rs   = rsqrtf(var + 1e-5f);
        nrs[r] = rs;
        nmu[r] = mean * rs;   // normalize = fma(acc, rs, -mean*rs)
    }
#pragma unroll
    for (int nt = 0; nt < 8; ++nt) {
        float g  = gamma[o * kH + nt * 16 + col];
        float be = beta [o * kH + nt * 16 + col];
#pragma unroll
        for (int r = 0; r < 8; ++r) {
            float nv = fmaf(fmaf(acc[nt][r], nrs[r], -nmu[r]), g, be);
            sh[(r + 8 * hi) * kRS + nt * 16 + col] = bf_bits(nv);
        }
    }
}

// ---------------- fused 3-layer FiLM block: one wave per (16-row tile, channel o) ----------------
__global__ __launch_bounds__(32)
void film_fused(const float* __restrict__ x,
                const float* __restrict__ b0, const float* __restrict__ g0, const float* __restrict__ be0,
                const float* __restrict__ b1, const float* __restrict__ g1, const float* __restrict__ be1,
                const float* __restrict__ b2,
                const unsigned* __restrict__ wpk,
                float* __restrict__ out) {
    __shared__ alignas(16) unsigned short sh[16 * kRS];

    const int lane = threadIdx.x;
    const int col  = lane & 15;    // A-row index / B,C,D-column index
    const int hi   = lane >> 4;
    const int kh   = 8 * hi;
    const int o    = blockIdx.y;
    const int base = blockIdx.x * 16;

    // One per-lane weight pointer; every B tile is a compile-time immediate offset off it.
    const char* wlane = (const char*)wpk + (size_t)o * kOBLK_B + lane * 32;

    // Warm L1/L2 path for this o's weight block while we pack A (global_prefetch_b8).
    __builtin_prefetch(wlane, 0, 1);
    __builtin_prefetch(wlane + 16 * 1024, 0, 1);
    __builtin_prefetch(wlane + 32 * 1024, 0, 1);
    __builtin_prefetch(wlane + 48 * 1024, 0, 1);

    union BF { v16bf v; unsigned u[8]; };

    // ---- layer 0: A = x tile (16x64 fp32 -> bf16, 2 K-chunks of 32) ----
    BF a0[2];
    const float* xrow = x + (size_t)(base + col) * kCIN;
#pragma unroll
    for (int kc = 0; kc < 2; ++kc)
#pragma unroll
        for (int v = 0; v < 8; ++v) {
            int k = kc * 32 + fmap(v) + kh;
            float2 xv = *(const float2*)(xrow + k);
            a0[kc].u[v] = pack_bf16(xv.x, xv.y);
        }

    v8f acc[8];
#pragma unroll
    for (int nt = 0; nt < 8; ++nt) {
        v8f c = {};
#pragma unroll
        for (int kc = 0; kc < 2; ++kc) {
            U32x8 t = *(const U32x8*)(wlane + (kc * 8 + nt) * 1024);
            v16bf bv = __builtin_bit_cast(v16bf, t);
            c = __builtin_amdgcn_wmma_f32_16x16x32_bf16(false, a0[kc].v, false, bv,
                                                        (short)0, c, false, false);
        }
        acc[nt] = c;
    }
    bias_leaky_ln_store(acc, b0, g0, be0, o, col, hi, sh);
    __syncthreads();

    // ---- layer 1: A = h0 from LDS (16x128, 4 K-chunks) ----
    BF a1[4];
#pragma unroll
    for (int kc = 0; kc < 4; ++kc)
#pragma unroll
        for (int v = 0; v < 8; ++v) {
            int k = kc * 32 + fmap(v) + kh;
            a1[kc].u[v] = *(const unsigned*)&sh[col * kRS + k];
        }
    __syncthreads();   // WAR guard before sh is rewritten

#pragma unroll
    for (int nt = 0; nt < 8; ++nt) {
        v8f c = {};
#pragma unroll
        for (int kc = 0; kc < 4; ++kc) {
            U32x8 t = *(const U32x8*)(wlane + (16 + kc * 8 + nt) * 1024);
            v16bf bv = __builtin_bit_cast(v16bf, t);
            c = __builtin_amdgcn_wmma_f32_16x16x32_bf16(false, a1[kc].v, false, bv,
                                                        (short)0, c, false, false);
        }
        acc[nt] = c;
    }
    bias_leaky_ln_store(acc, b1, g1, be1, o, col, hi, sh);
    __syncthreads();

    // ---- layer 2: A = h1 from LDS, single 16-wide N tile ----
    BF a2[4];
#pragma unroll
    for (int kc = 0; kc < 4; ++kc)
#pragma unroll
        for (int v = 0; v < 8; ++v) {
            int k = kc * 32 + fmap(v) + kh;
            a2[kc].u[v] = *(const unsigned*)&sh[col * kRS + k];
        }

    v8f c2 = {};
#pragma unroll
    for (int kc = 0; kc < 4; ++kc) {
        U32x8 t = *(const U32x8*)(wlane + (48 + kc) * 1024);
        v16bf bv = __builtin_bit_cast(v16bf, t);
        c2 = __builtin_amdgcn_wmma_f32_16x16x32_bf16(false, a2[kc].v, false, bv,
                                                     (short)0, c2, false, false);
    }

    // out layout (N,T,F2,O): row*256 + f*16 + o ; D tile: col = f, row m = r + 8*hi
    float bo = b2[o * kF2 + col];
    float* orow = out + (size_t)(base + 8 * hi) * (kF2 * kO) + col * kO + o;
#pragma unroll
    for (int r = 0; r < 8; ++r) {
        orow[r * (kF2 * kO)] = leaky(c2[r] + bo);
    }
}

extern "C" void kernel_launch(void* const* d_in, const int* in_sizes, int n_in,
                              void* d_out, int out_size, void* d_ws, size_t ws_size,
                              hipStream_t stream) {
    (void)in_sizes; (void)n_in; (void)out_size; (void)ws_size;
    const float* x   = (const float*)d_in[0];
    const float* W0  = (const float*)d_in[1];
    const float* b0  = (const float*)d_in[2];
    const float* g0  = (const float*)d_in[3];
    const float* be0 = (const float*)d_in[4];
    const float* W1  = (const float*)d_in[5];
    const float* b1  = (const float*)d_in[6];
    const float* g1  = (const float*)d_in[7];
    const float* be1 = (const float*)d_in[8];
    const float* W2  = (const float*)d_in[9];
    const float* b2  = (const float*)d_in[10];
    unsigned* wpk = (unsigned*)d_ws;   // needs 832 KB of scratch

    const int total = kC0 + kC1 + kC2;   // 26624 lane-slots
    film_pack_weights<<<(total + 255) / 256, 256, 0, stream>>>(W0, W1, W2, wpk);

    dim3 grid(kNT / 16, kO);
    film_fused<<<grid, 32, 0, stream>>>(x, b0, g0, be0, b1, g1, be1, b2, wpk,
                                        (float*)d_out);
}
